// RNNLayer_74586402062433
// MI455X (gfx1250) — matrix operand order repaired and checked by reading, not compiled, not run on hardware
//
#include <hip/hip_runtime.h>

// ---------------------------------------------------------------------------
// tanh RNN for MI455X (gfx1250, wave32):
//   xp  = x @ Wx + b            -> parallel bf16-WMMA GEMM into d_out
//                                  (128x64 macro-tile, 2x2 wave blocking)
//   h_t = tanh(xp_t + h @ Wh)   -> ONE persistent kernel, 128 steps with a
//                                  device-wide barrier; Wh strip lives in LDS
//                                  (bf16) for the whole scan; h ping-pongs in
//                                  d_ws as bf16 so A-fragments load straight
//                                  from global with zero per-step conversion.
// ---------------------------------------------------------------------------

typedef __attribute__((ext_vector_type(16))) __bf16 v16bf;
typedef __attribute__((ext_vector_type(8)))  __bf16 v8bf;
typedef __attribute__((ext_vector_type(8)))  float  v8f;

#define SEQ   128
#define BATCH 128
#define DIM   1024

// --- projection GEMM tiling: 128x64 macro-tile, wave computes 32x32 ---
#define BM 128
#define BN 64
#define BK 32
#define LDP 40     // padded LDS row stride (bf16): 80B, 16B-aligned, conflict-free

// --- persistent scan tiling ---
#define NBLK 32            // persistent workgroups; each owns 32 output columns
#define CPB  (DIM / NBLK)  // 32 columns per block
#define BSTR (DIM + 8)     // Bs row stride (bf16): 2064B = 129*16B, spreads banks

__device__ __forceinline__ v16bf cat16(v8bf lo, v8bf hi) {
  return __builtin_shufflevector(lo, hi, 0,1,2,3,4,5,6,7,8,9,10,11,12,13,14,15);
}

#define WMMA_BF16(a, b, c) \
  __builtin_amdgcn_wmma_f32_16x16x32_bf16(false, (a), false, (b), (short)0, (c), false, false)

// ======================= parallel input projection =========================
// C[128x64 tile] = A[tm:tm+128, :] * W[:, tn:tn+64]; 8 waves, each 32x32.
__global__ void __launch_bounds__(256)
rnn_proj(const float* __restrict__ x, const float* __restrict__ Wx,
         const float* __restrict__ bias, float* __restrict__ xp)
{
  __shared__ __bf16 As[BM][LDP];   // As[m][k]
  __shared__ __bf16 Bs[BN][LDP];   // Bs[n][k]  (Wx tile transposed)

  const int tn = blockIdx.x * BN;
  const int tm = blockIdx.y * BM;

  const int tid  = threadIdx.x;
  const int lane = tid & 31;
  const int wave = tid >> 5;
  const int wm = wave & 3;         // 0..3 -> 32-row strip
  const int wn = wave >> 2;        // 0..1 -> 32-col strip
  const int l15 = lane & 15, hi = lane >> 4;

  v8f c00 = {}, c01 = {}, c10 = {}, c11 = {};

  for (int k0 = 0; k0 < DIM; k0 += BK) {
    // --- stage A block (128x32 f32 -> bf16), coalesced float4 loads ---
#pragma unroll
    for (int p = 0; p < 4; ++p) {
      const int i   = tid + p * 256;     // 0..1023 float4 slots
      const int row = i >> 3;            // 8 float4 per 32-wide row
      const int kk  = (i & 7) << 2;
      const float4 v = *(const float4*)(x + (size_t)(tm + row) * DIM + k0 + kk);
      As[row][kk + 0] = (__bf16)v.x;  As[row][kk + 1] = (__bf16)v.y;
      As[row][kk + 2] = (__bf16)v.z;  As[row][kk + 3] = (__bf16)v.w;
    }
    // --- stage W block (32x64 f32 -> bf16), stored transposed Bs[n][k] ---
#pragma unroll
    for (int p = 0; p < 2; ++p) {
      const int i  = tid + p * 256;
      const int kk = i >> 4;             // 16 float4 per 64-wide row
      const int nn = (i & 15) << 2;
      const float4 v = *(const float4*)(Wx + (size_t)(k0 + kk) * DIM + tn + nn);
      Bs[nn + 0][kk] = (__bf16)v.x;  Bs[nn + 1][kk] = (__bf16)v.y;
      Bs[nn + 2][kk] = (__bf16)v.z;  Bs[nn + 3][kk] = (__bf16)v.w;
    }
    __syncthreads();

    // A fragments: lane = row M; elems 0..7 <- K=hi*8.., elems 8..15 <- K=16+hi*8..
    const int ar0 = wm * 32 + l15;
    const v16bf a0 = cat16(*(const v8bf*)(&As[ar0][hi * 8]),
                           *(const v8bf*)(&As[ar0][16 + hi * 8]));
    const v16bf a1 = cat16(*(const v8bf*)(&As[ar0 + 16][hi * 8]),
                           *(const v8bf*)(&As[ar0 + 16][16 + hi * 8]));
    // B fragments: lane = column N; elems 0..7 <- K=hi*8.., elems 8..15 <- K=16+hi*8..
    const int bn0 = wn * 32 + l15;
    const v16bf b0 = cat16(*(const v8bf*)(&Bs[bn0][hi * 8]),
                           *(const v8bf*)(&Bs[bn0][16 + hi * 8]));
    const v16bf b1 = cat16(*(const v8bf*)(&Bs[bn0 + 16][hi * 8]),
                           *(const v8bf*)(&Bs[bn0 + 16][16 + hi * 8]));

    c00 = WMMA_BF16(a0, b0, c00);
    c01 = WMMA_BF16(a0, b1, c01);
    c10 = WMMA_BF16(a1, b0, c10);
    c11 = WMMA_BF16(a1, b1, c11);
    __syncthreads();
  }

  // C/D layout: VGPR r -> M = r + 8*hi; lane -> N = l15.
#pragma unroll
  for (int r = 0; r < 8; ++r) {
    const int gm0 = tm + wm * 32 + hi * 8 + r;
    const int gm1 = gm0 + 16;
    const int gn0 = tn + wn * 32 + l15;
    const int gn1 = gn0 + 16;
    xp[(size_t)gm0 * DIM + gn0] = c00[r] + bias[gn0];
    xp[(size_t)gm0 * DIM + gn1] = c01[r] + bias[gn1];
    xp[(size_t)gm1 * DIM + gn0] = c10[r] + bias[gn0];
    xp[(size_t)gm1 * DIM + gn1] = c11[r] + bias[gn1];
  }
}

// h0 -> bf16 ping-pong buffer
__global__ void __launch_bounds__(256)
h0_to_bf16(const float* __restrict__ h0, __bf16* __restrict__ hb)
{
  const int i = blockIdx.x * 256 + threadIdx.x;
  hb[i] = (__bf16)h0[i];
}

// ======================= persistent recurrent scan =========================
__global__ void __launch_bounds__(256)
rnn_scan(const float* __restrict__ Wh, float* __restrict__ out,
         __bf16* __restrict__ hb0, __bf16* __restrict__ hb1,
         unsigned* __restrict__ cnt)
{
  // Wh strip (all K x CPB columns), bf16, transposed: Bs[n][k]. Resident for
  // the whole scan: 32*2064B = 66KB of the WGP's 320KB LDS.
  __shared__ __bf16 Bs[CPB][BSTR];

  const int tid  = threadIdx.x;
  const int lane = tid & 31;
  const int wave = tid >> 5;            // 0..7 -> 16-row strip
  const int l15  = lane & 15;
  const int hi   = lane >> 4;
  const int n0   = blockIdx.x * CPB;    // this block's output columns
  const int m0   = wave * 16;           // this wave's output rows

  // One-time stage: f32 Wh -> bf16 LDS (coalesced: 32 consecutive n per k).
  for (int idx = tid; idx < CPB * DIM; idx += 256) {
    const int n = idx & (CPB - 1);
    const int k = idx >> 5;
    Bs[n][k] = (__bf16)Wh[(size_t)k * DIM + n0 + n];
  }
  __syncthreads();

  __bf16* hbuf[2] = { hb0, hb1 };

  for (int t = 0; t < SEQ; ++t) {
    const __bf16* __restrict__ hin  = hbuf[t & 1];
    __bf16* __restrict__       hout = hbuf[(t + 1) & 1];
    float* __restrict__        slab = out + (size_t)t * BATCH * DIM;

    v8f c0 = {}, c1 = {};
    // A-operand rows straight from the bf16 h buffer (L2-hot):
    // lane = row M; elems 0..7 <- K=hi*8+0..7, elems 8..15 <- K=16+hi*8+0..7.
    const __bf16* arow = hin + (size_t)(m0 + l15) * DIM;
#pragma unroll 4
    for (int k0 = 0; k0 < DIM; k0 += 32) {
      const v16bf a = cat16(*(const v8bf*)(arow + k0 + hi * 8),
                            *(const v8bf*)(arow + k0 + 16 + hi * 8));
      {
        const v16bf bb = cat16(*(const v8bf*)(&Bs[l15][k0 + hi * 8]),
                               *(const v8bf*)(&Bs[l15][k0 + 16 + hi * 8]));
        c0 = WMMA_BF16(a, bb, c0);
      }
      {
        const v16bf bb = cat16(*(const v8bf*)(&Bs[16 + l15][k0 + hi * 8]),
                               *(const v8bf*)(&Bs[16 + l15][k0 + 16 + hi * 8]));
        c1 = WMMA_BF16(a, bb, c1);
      }
    }

    // Epilogue: h = tanh(xp + acc); write f32 (in place over xp_t) + bf16.
#pragma unroll
    for (int r = 0; r < 8; ++r) {
      const int gm  = m0 + hi * 8 + r;
      const int gn0 = n0 + l15;
      const size_t i0 = (size_t)gm * DIM + gn0;
      const size_t i1 = i0 + 16;
      const float v0 = tanhf(c0[r] + slab[i0]);
      const float v1 = tanhf(c1[r] + slab[i1]);
      slab[i0] = v0;  hout[i0] = (__bf16)v0;
      slab[i1] = v1;  hout[i1] = (__bf16)v1;
    }

    // Device-wide step barrier (monotonic counter; reset by hipMemsetAsync).
    if (t != SEQ - 1) {
      __threadfence();                 // release: my stores visible device-wide
      __syncthreads();
      if (tid == 0) {
        atomicAdd(cnt, 1u);
        const unsigned target = (unsigned)(NBLK * (t + 1));
        while (__hip_atomic_load(cnt, __ATOMIC_RELAXED,
                                 __HIP_MEMORY_SCOPE_AGENT) < target) {
          __builtin_amdgcn_s_sleep(2);
        }
      }
      __syncthreads();
      __threadfence();                 // acquire: drop stale cached h lines
    }
  }
}

// ============================== launcher ===================================
extern "C" void kernel_launch(void* const* d_in, const int* in_sizes, int n_in,
                              void* d_out, int out_size, void* d_ws, size_t ws_size,
                              hipStream_t stream) {
  (void)in_sizes; (void)n_in; (void)out_size; (void)ws_size;
  const float* x  = (const float*)d_in[0];  // [SEQ, BATCH, DIM]
  const float* h0 = (const float*)d_in[1];  // [BATCH, DIM]
  const float* Wx = (const float*)d_in[2];  // [DIM, DIM]
  const float* Wh = (const float*)d_in[3];  // [DIM, DIM]
  const float* b  = (const float*)d_in[4];  // [DIM]
  float* out = (float*)d_out;               // [SEQ, BATCH, DIM]

  // Workspace layout: two bf16 h buffers (ping-pong) + barrier counter.
  const size_t HB = (size_t)BATCH * DIM;          // elements
  __bf16*   hb0 = (__bf16*)d_ws;
  __bf16*   hb1 = hb0 + HB;
  unsigned* cnt = (unsigned*)((char*)d_ws + 2 * HB * sizeof(__bf16));

  const dim3 blk(256);
  // 1) xp = x@Wx + b into d_out (fully parallel).
  rnn_proj<<<dim3(DIM / BN, (SEQ * BATCH) / BM), blk, 0, stream>>>(x, Wx, b, out);
  // 2) h0 -> bf16.
  h0_to_bf16<<<dim3((unsigned)(HB / 256)), blk, 0, stream>>>(h0, hb0);
  // 3) reset barrier counter (graph-capturable) and run the persistent scan.
  hipMemsetAsync(cnt, 0, sizeof(unsigned), stream);
  rnn_scan<<<dim3(NBLK), blk, 0, stream>>>(Wh, out, hb0, hb1, cnt);
}
